// DariushMoELayer_14087492731057
// MI455X (gfx1250) — matrix operand order repaired and checked by reading, not compiled, not run on hardware
//
#include <hip/hip_runtime.h>
#include <stdint.h>

// ---------------- problem constants (B=4, S=2048, D=512, E=16, top-2) -------
#define T_TOKENS 8192
#define D_EMB    512
#define N_EXP    16
#define MT       128                  // token tile per FFN block
#define TILES    (T_TOKENS / MT)      // 64

typedef __attribute__((ext_vector_type(16))) __bf16 v16bf;
typedef __attribute__((ext_vector_type(8)))  __bf16 v8bf;
typedef __attribute__((ext_vector_type(8)))  float  v8f;

// ---------------- helpers ----------------------------------------------------
__device__ __forceinline__ float gumbel_noise(uint32_t i) {
    // deterministic hash -> U(0,1) -> Gumbel. (Exact JAX threefry stream is not
    // reproducible here; only affects near-tie top-k picks.)
    uint32_t s = i * 0x9E3779B9u + 0x85EBCA6Bu;
    s ^= s >> 16; s *= 0x7FEB352Du;
    s ^= s >> 15; s *= 0x846CA68Bu;
    s ^= s >> 16;
    float u = ((float)(s >> 8) + 0.5f) * (1.0f / 16777216.0f);
    return -__logf(-__logf(u));
}

// ---------------- kernel 1: zero output + expert counters --------------------
__global__ void k_zero(float4* out4, int n4, int* cnt) {
    int i = blockIdx.x * blockDim.x + threadIdx.x;
    if (i < n4) out4[i] = make_float4(0.f, 0.f, 0.f, 0.f);
    if (i < N_EXP) cnt[i] = 0;
}

// ---------------- kernel 2: fp32 -> bf16 conversion ---------------------------
__global__ void k_cvt(const float* __restrict__ src, __bf16* __restrict__ dst, int n) {
    int i = blockIdx.x * blockDim.x + threadIdx.x;
    if (i < n) dst[i] = (__bf16)src[i];
}

// ---------------- kernel 3: router (1 wave per token) -------------------------
__global__ void k_router(const float* __restrict__ x, const float* __restrict__ wr,
                         float* __restrict__ gates, int* __restrict__ idx) {
    int token = blockIdx.x * 4 + (threadIdx.x >> 5);
    int lane  = threadIdx.x & 31;
    const float* xr = x + (size_t)token * D_EMB;

    float acc[N_EXP];
#pragma unroll
    for (int e = 0; e < N_EXP; ++e) acc[e] = 0.f;

    for (int d = lane; d < D_EMB; d += 32) {
        float xv = xr[d];
        const float* w = wr + d * N_EXP;
#pragma unroll
        for (int e = 0; e < N_EXP; ++e) acc[e] += xv * w[e];
    }
#pragma unroll
    for (int off = 16; off >= 1; off >>= 1)
#pragma unroll
        for (int e = 0; e < N_EXP; ++e) acc[e] += __shfl_xor(acc[e], off, 32);

    if (lane == 0) {
        float z[N_EXP];
        float m = -1e30f;
#pragma unroll
        for (int e = 0; e < N_EXP; ++e) {
            z[e] = acc[e] + 0.05f * gumbel_noise((uint32_t)(token * N_EXP + e));
            m = fmaxf(m, z[e]);
        }
        float s = 0.f;
#pragma unroll
        for (int e = 0; e < N_EXP; ++e) { z[e] = __expf(z[e] - m); s += z[e]; }
        int i0 = 0; float p0 = -1.f;
#pragma unroll
        for (int e = 0; e < N_EXP; ++e) if (z[e] > p0) { p0 = z[e]; i0 = e; }
        int i1 = 0; float p1 = -1.f;
#pragma unroll
        for (int e = 0; e < N_EXP; ++e) if (e != i0 && z[e] > p1) { p1 = z[e]; i1 = e; }
        float inv = 1.f / s;
        gates[token * 2 + 0] = p0 * inv;
        gates[token * 2 + 1] = p1 * inv;
        idx[token * 2 + 0] = i0;
        idx[token * 2 + 1] = i1;
    }
}

// ---------------- kernel 4: bucket tokens by expert ---------------------------
__global__ void k_bucket(const int* __restrict__ idx, int* __restrict__ cnt,
                         int* __restrict__ lists) {
    int t = blockIdx.x * blockDim.x + threadIdx.x;
    if (t >= T_TOKENS) return;
#pragma unroll
    for (int k = 0; k < 2; ++k) {
        int e = idx[t * 2 + k];
        int slot = atomicAdd(&cnt[e], 1);
        lists[e * T_TOKENS + slot] = t * 2 + k;   // encode token*2 + k-slot
    }
}

// ---------------- kernel 5: fused SwiGLU expert FFN (bf16 WMMA) ---------------
// LDS: X tile 128x512 bf16 (128KB) + U tile 128x512 bf16 (128KB)
//      + two 16x512 weight slabs (32KB) + token/gate arrays (1KB)  -> 289KB/WGP
__global__ __launch_bounds__(256) void k_ffn(
    const __bf16* __restrict__ xb,
    const __bf16* __restrict__ w1, const __bf16* __restrict__ w2,
    const __bf16* __restrict__ wo,
    const float* __restrict__ b1, const float* __restrict__ b2,
    const float* __restrict__ bo,
    const float* __restrict__ gates, const int* __restrict__ lists,
    const int* __restrict__ cnt, float* __restrict__ out)
{
    extern __shared__ char smem[];
    __bf16* sX   = (__bf16*)smem;              // MT * 512
    __bf16* sU   = sX + MT * D_EMB;            // MT * 512
    __bf16* sW   = sU + MT * D_EMB;            // 2 * 16 * 512 (transposed slabs)
    int*    sTok = (int*)(sW + 2 * 16 * D_EMB);
    float*  sGate = (float*)(sTok + MT);

    const int e    = blockIdx.x / TILES;
    const int tile = blockIdx.x % TILES;
    const int ne   = cnt[e];
    if (tile * MT >= ne) return;               // inactive tile
    const int base = tile * MT;
    const int rows = (ne - base < MT) ? (ne - base) : MT;

    const int tid  = threadIdx.x;
    const int wv   = tid >> 5;
    const int lane = tid & 31;

    // token ids + gates for this tile
    for (int r = tid; r < MT; r += 256) {
        if (r < rows) {
            int enc = lists[e * T_TOKENS + base + r];
            sTok[r]  = enc >> 1;
            sGate[r] = gates[enc];
        } else { sTok[r] = -1; sGate[r] = 0.f; }
    }
    __syncthreads();

    // ---- gather x rows into LDS via CDNA5 async global->LDS copies ----------
    // Row-major copy (no transpose) -> GLOBAL_LOAD_ASYNC_TO_LDS_B128, tracked
    // on ASYNCcnt; drained once with s_wait_asynccnt before the barrier.
    {
        const uint32_t sX_lds = (uint32_t)(uintptr_t)sX;   // LDS byte offset
        for (int c = tid; c < MT * (D_EMB / 8); c += 256) {
            int r = c >> 6, ch = c & 63;
            int tok = sTok[r];
            if (tok >= 0) {
                uint32_t lds_off = sX_lds + (uint32_t)(r * (D_EMB * 2) + ch * 16);
                uint64_t gaddr   = (uint64_t)(uintptr_t)(xb + (size_t)tok * D_EMB) +
                                   (uint64_t)(ch * 16);
                asm volatile("global_load_async_to_lds_b128 %0, %1, off"
                             :: "v"(lds_off), "v"(gaddr) : "memory");
            } else {
                uint4 z = make_uint4(0u, 0u, 0u, 0u);
                *(uint4*)(sX + (size_t)r * D_EMB + ch * 8) = z;
            }
        }
        asm volatile("s_wait_asynccnt 0x0" ::: "memory");
    }

    // per-wave fragment geometry (contiguous k order: lane half h holds k = 16h..16h+15)
    const int R     = wv * 16;                 // wave's 16-row strip
    const int mA    = R + (lane & 15);
    const int koffA = (lane < 16) ? 0 : 16;    // A: K0..15 vs K16..31
    const int nB    = lane & 15;
    const int koffB = (lane < 16) ? 0 : 16;    // B: K0..15 vs K16..31
    const int nl    = lane & 15;
    const int mofs  = (lane < 16) ? 0 : 8;
    const __bf16* w1g = w1 + (size_t)e * D_EMB * D_EMB;
    const __bf16* w2g = w2 + (size_t)e * D_EMB * D_EMB;
    const __bf16* wog = wo + (size_t)e * D_EMB * D_EMB;

    // ---------------- phase 1: h1 = X@W1+b1, h2 = X@W2+b2, U = silu(h1)*h2 ---
    for (int nt = 0; nt < D_EMB / 16; ++nt) {
        const int ncol = nt * 16;
        __syncthreads();                       // protect previous slab reads
        // stage W1/W2 column slabs transposed via vector global loads:
        // load 8 contiguous bf16 (B128) of row k, scatter into sW[n][k].
        for (int c = tid; c < 2048; c += 256) {
            int mat  = c >> 10;                // 0 = W1, 1 = W2
            int j    = c & 1023;
            int k    = j >> 1;
            int half = (j & 1) << 3;           // column sub-offset 0 or 8
            const __bf16* g = (mat ? w2g : w1g) + (size_t)k * D_EMB + ncol + half;
            v8bf v = *(const v8bf*)g;
            __bf16* dstb = sW + mat * 16 * D_EMB;
#pragma unroll
            for (int t = 0; t < 8; ++t) dstb[(half + t) * D_EMB + k] = v[t];
        }
        __syncthreads();

        v8f c1 = {}; v8f c2 = {};
#pragma unroll 4
        for (int ks = 0; ks < 16; ++ks) {
            const int k0 = ks * 32;
            union { v16bf v; v8bf h[2]; } a;
            a.h[0] = *(const v8bf*)(sX + (size_t)mA * D_EMB + k0 + koffA);
            a.h[1] = *(const v8bf*)(sX + (size_t)mA * D_EMB + k0 + koffA + 8);
            v16bf bb1 = *(const v16bf*)(sW + (size_t)nB * D_EMB + k0 + koffB);
            v16bf bb2 = *(const v16bf*)(sW + 16 * D_EMB + (size_t)nB * D_EMB + k0 + koffB);
            c1 = __builtin_amdgcn_wmma_f32_16x16x32_bf16(false, a.v, false, bb1,
                                                         (short)0, c1, false, false);
            c2 = __builtin_amdgcn_wmma_f32_16x16x32_bf16(false, a.v, false, bb2,
                                                         (short)0, c2, false, false);
        }
#pragma unroll
        for (int i = 0; i < 8; ++i) {
            int m = R + i + mofs;
            int n = ncol + nl;
            float h1 = c1[i] + b1[(e << 9) + n];
            float h2 = c2[i] + b2[(e << 9) + n];
            float uu = (h1 / (1.f + __expf(-h1))) * h2;   // silu(h1) * h2
            sU[(size_t)m * D_EMB + n] = (__bf16)uu;
        }
    }

    // ---------------- phase 2: Y = U@Wout + bout, out += gate*Y (atomic) -----
    for (int nt = 0; nt < D_EMB / 16; ++nt) {
        const int ncol = nt * 16;
        __syncthreads();                       // U complete / previous slab reads done
        for (int c = tid; c < 1024; c += 256) {
            int k    = c >> 1;
            int half = (c & 1) << 3;
            const __bf16* g = wog + (size_t)k * D_EMB + ncol + half;
            v8bf v = *(const v8bf*)g;
#pragma unroll
            for (int t = 0; t < 8; ++t) sW[(half + t) * D_EMB + k] = v[t];
        }
        __syncthreads();

        v8f c = {};
#pragma unroll 4
        for (int ks = 0; ks < 16; ++ks) {
            const int k0 = ks * 32;
            union { v16bf v; v8bf h[2]; } a;
            a.h[0] = *(const v8bf*)(sU + (size_t)mA * D_EMB + k0 + koffA);
            a.h[1] = *(const v8bf*)(sU + (size_t)mA * D_EMB + k0 + koffA + 8);
            v16bf bb = *(const v16bf*)(sW + (size_t)nB * D_EMB + k0 + koffB);
            c = __builtin_amdgcn_wmma_f32_16x16x32_bf16(false, a.v, false, bb,
                                                        (short)0, c, false, false);
        }
#pragma unroll
        for (int i = 0; i < 8; ++i) {
            int m = R + i + mofs;
            int n = ncol + nl;
            int tok = sTok[m];
            if (tok >= 0) {
                float y = c[i] + bo[(e << 9) + n];
                unsafeAtomicAdd(&out[(size_t)tok * D_EMB + n], sGate[m] * y);
            }
        }
    }
}

// ---------------- host launcher ----------------------------------------------
extern "C" void kernel_launch(void* const* d_in, const int* in_sizes, int n_in,
                              void* d_out, int out_size, void* d_ws, size_t ws_size,
                              hipStream_t stream) {
    const float* x    = (const float*)d_in[0];
    const float* wr   = (const float*)d_in[1];
    const float* W1   = (const float*)d_in[2];
    const float* b1   = (const float*)d_in[3];
    const float* W2   = (const float*)d_in[4];
    const float* b2   = (const float*)d_in[5];
    const float* Wout = (const float*)d_in[6];
    const float* bout = (const float*)d_in[7];
    float* out = (float*)d_out;

    // workspace layout (bytes)
    char* ws = (char*)d_ws;
    const size_t NX = (size_t)T_TOKENS * D_EMB;          // 4,194,304
    const size_t NW = (size_t)N_EXP * D_EMB * D_EMB;     // 4,194,304
    __bf16* xb   = (__bf16*)(ws);
    __bf16* w1b  = (__bf16*)(ws + NX * 2);
    __bf16* w2b  = (__bf16*)(ws + NX * 2 + NW * 2);
    __bf16* wob  = (__bf16*)(ws + NX * 2 + NW * 4);
    char*   tail = ws + NX * 2 + NW * 6;
    float*  gates = (float*)tail;                         // T*2 floats
    int*    idx   = (int*)(tail + T_TOKENS * 2 * 4);      // T*2 ints
    int*    cnt   = (int*)(tail + T_TOKENS * 4 * 4);      // 16 ints (padded)
    int*    lists = (int*)(tail + T_TOKENS * 4 * 4 + 256);// E * T ints

    // 1) zero output (as float4) + counters
    int n4 = (int)(NX / 4);
    k_zero<<<(n4 + 255) / 256, 256, 0, stream>>>((float4*)out, n4, cnt);

    // 2) fp32 -> bf16
    k_cvt<<<(int)((NX + 255) / 256), 256, 0, stream>>>(x,    xb,  (int)NX);
    k_cvt<<<(int)((NW + 255) / 256), 256, 0, stream>>>(W1,   w1b, (int)NW);
    k_cvt<<<(int)((NW + 255) / 256), 256, 0, stream>>>(W2,   w2b, (int)NW);
    k_cvt<<<(int)((NW + 255) / 256), 256, 0, stream>>>(Wout, wob, (int)NW);

    // 3) router: 4 tokens per 128-thread block (1 wave/token)
    k_router<<<T_TOKENS / 4, 128, 0, stream>>>(x, wr, gates, idx);

    // 4) bucket by expert
    k_bucket<<<T_TOKENS / 256, 256, 0, stream>>>(idx, cnt, lists);

    // 5) fused expert FFN, 289KB dynamic LDS per workgroup (320KB WGP limit)
    size_t shmem = (size_t)MT * D_EMB * 2 * 2      // sX + sU
                 + (size_t)2 * 16 * D_EMB * 2      // weight slabs
                 + (size_t)MT * (4 + 4);           // tok + gate
    (void)hipFuncSetAttribute((const void*)k_ffn,
                              hipFuncAttributeMaxDynamicSharedMemorySize, (int)shmem);
    k_ffn<<<N_EXP * TILES, 256, shmem, stream>>>(xb, w1b, w2b, wob,
                                                 b1, b2, bout,
                                                 gates, lists, cnt, out);
}
